// NewVectorQuantizer_41154376630736
// MI455X (gfx1250) — compile-verified, matrix operand order checked by reading, not run on hardware
//
#include <hip/hip_runtime.h>

typedef __bf16 bf16_t;
typedef __attribute__((ext_vector_type(16))) __bf16 v16bf;
typedef __attribute__((ext_vector_type(8)))  __bf16 v8bf;
typedef __attribute__((ext_vector_type(8)))  float  v8f;

#define D_DIM   256
#define K_DIM   1024
#define HW      4096          // 64*64
#define N_TOTAL 131072        // 32*64*64
#define N_TILE  128
#define A_STRIDE 264          // 256 + 8 bf16 pad (16B) to spread LDS banks
#define OUT_PLANE 33554432    // 32*256*64*64
#define KB_STEPS 64           // K_DIM / 16

// ---------------------------------------------------------------------------
// Prep 1: repack embed [D,K] f32 -> bf16 WMMA B-fragment layout.
// Fragment (kb, t) covers codes kb*16..kb*16+15 (N) and dims t*32..t*32+31 (K).
// Per ISA 7.12.2 (16-bit operands): lane L -> column N = L%16;
//   elements 0..7  = K { half*8 + 0..7 }        (half = L/16)
//   elements 8..15 = K { 16 + half*8 + 0..7 }
// ---------------------------------------------------------------------------
__global__ __launch_bounds__(256) void prep_b_kernel(const float* __restrict__ embed,
                                                     bf16_t* __restrict__ bprep) {
    int gid  = blockIdx.x * 256 + threadIdx.x;   // 0..16383  (512 frags * 32 lanes)
    int lane = gid & 31;
    int frag = gid >> 5;                         // 0..511
    int kb   = frag >> 3;                        // 0..63
    int t    = frag & 7;                         // 0..7
    int k    = kb * 16 + (lane & 15);
    int base_d = t * 32 + (lane >> 4) * 8;
    v16bf o;
#pragma unroll
    for (int j = 0; j < 8; ++j) {
        o[j]     = (bf16_t)embed[(base_d + j) * K_DIM + k];
        o[j + 8] = (bf16_t)embed[(base_d + 16 + j) * K_DIM + k];
    }
    ((v16bf*)bprep)[gid] = o;
}

// Prep 2: ||e_k||^2 per code, exact f32. Coalesced: thread k strides down D.
__global__ __launch_bounds__(256) void enorm_kernel(const float* __restrict__ embed,
                                                    float* __restrict__ enorm) {
    int k = blockIdx.x * 256 + threadIdx.x;      // 0..1023
    float s = 0.0f;
    for (int d = 0; d < D_DIM; ++d) {
        float v = embed[d * K_DIM + k];
        s = __builtin_fmaf(v, v, s);
    }
    enorm[k] = s;
}

// ---------------------------------------------------------------------------
// Main: per workgroup, 128 contiguous pixels x full K=1024 codebook.
// score[n,k] = ||e_k||^2 - 2 * (x_n . e_k); argmin over k.
// B fragments double-buffered through LDS (shared by all 8 waves).
// ---------------------------------------------------------------------------
__global__ __launch_bounds__(256) void vq_main(const float* __restrict__ input,
                                               const float* __restrict__ embed,
                                               const bf16_t* __restrict__ bprep,
                                               const float* __restrict__ enorm,
                                               float* __restrict__ out) {
    __shared__ bf16_t Atile[N_TILE * A_STRIDE];       // 67.5 KB
    __shared__ bf16_t Bbuf[2 * 256 * 16];             // 16 KB: 2 x (8 frags x 1KB)
    __shared__ float  enorm_lds[K_DIM];               // 4 KB
    __shared__ int    idx_lds[N_TILE];

    const int tid  = threadIdx.x;
    const int lane = tid & 31;
    const int wave = tid >> 5;                   // 0..7, wave owns rows 16w..16w+15
    const int n0   = blockIdx.x * N_TILE;        // global flat pixel base
    const int b    = n0 / HW;
    const int off  = n0 % HW;                    // tile never crosses a batch (128|4096)
    const size_t in_base = (size_t)b * D_DIM * HW + off;

    const v16bf* BG = (const v16bf*)bprep;       // global, v16bf granularity (32B)
    v16bf*       BL = (v16bf*)Bbuf;              // LDS view

    // ---- Stage A tile (bf16, [pixel][d]), enorm, and B-step 0 into LDS
    {
        const int p  = tid & 127;                // pixel in tile
        const int dp = tid >> 7;                 // 0/1: d parity slot
#pragma unroll 4
        for (int it = 0; it < 128; ++it) {
            int d = it * 2 + dp;
            float v = __builtin_nontemporal_load(&input[in_base + (size_t)d * HW + p]);
            Atile[p * A_STRIDE + d] = (bf16_t)v;
        }
        // enorm: 4 floats per thread, coalesced
        ((float4*)enorm_lds)[tid] = ((const float4*)enorm)[tid];
        // B fragments for kb = 0: 8 KB = 256 threads x 32 B
        BL[tid] = BG[tid];
    }
    __syncthreads();

    // ---- Build 8 resident A fragments (16 rows x 32 dims each) per wave.
    const int halfw = lane >> 4;
    const int row   = (wave << 4) + (lane & 15);
    v16bf afrag[8];
#pragma unroll
    for (int t = 0; t < 8; ++t) {
        const v8bf* p0 = (const v8bf*)&Atile[row * A_STRIDE + t * 32 + halfw * 8];
        v8bf lo = p0[0];
        v8bf hi = p0[2];                         // +16 bf16 elements
        afrag[t] = __builtin_shufflevector(lo, hi, 0,1,2,3,4,5,6,7,8,9,10,11,12,13,14,15);
    }

    float best[8];
    int   bidx[8];
#pragma unroll
    for (int g = 0; g < 8; ++g) { best[g] = 3.4e38f; bidx[g] = 0; }

#pragma unroll 1
    for (int kb = 0; kb < KB_STEPS; ++kb) {
        const int cur = kb & 1;
        // Issue next step's global fetch early; its wait lands at the ds_store
        // below, hidden behind 8 WMMAs + ds reads.
        v16bf stagereg;
        const bool has_next = (kb + 1) < KB_STEPS;
        if (has_next) stagereg = BG[(kb + 1) * 256 + tid];

        float en = enorm_lds[kb * 16 + (lane & 15)];

        // Rotate-prefetch B fragments from LDS: load t+1 before WMMA on t.
        v8f acc = {};
        v16bf bc = BL[cur * 256 + lane];         // frag t = 0
#pragma unroll
        for (int t = 0; t < 8; ++t) {
            v16bf bn = BL[cur * 256 + (((t + 1) & 7) << 5) + lane];
            acc = __builtin_amdgcn_wmma_f32_16x16x32_bf16(
                false, afrag[t], false, bc, (short)0, acc, false, false);
            bc = bn;
        }

        // C/D layout: lane L -> column N = L%16, VGPR g -> row M = g + 8*(L/16)
        int kidx = kb * 16 + (lane & 15);
#pragma unroll
        for (int g = 0; g < 8; ++g) {
            float s = __builtin_fmaf(-2.0f, acc[g], en);
            if (s < best[g]) { best[g] = s; bidx[g] = kidx; }
        }

        if (has_next) BL[(1 - cur) * 256 + tid] = stagereg;
        __syncthreads();
    }

    // ---- Cross-lane argmin across the 16 lanes sharing the same row set
#pragma unroll
    for (int m = 8; m >= 1; m >>= 1) {
#pragma unroll
        for (int g = 0; g < 8; ++g) {
            float s2 = __shfl_xor(best[g], m, 32);
            int   i2 = __shfl_xor(bidx[g], m, 32);
            if (s2 < best[g] || (s2 == best[g] && i2 < bidx[g])) {
                best[g] = s2; bidx[g] = i2;
            }
        }
    }
    if ((lane & 15) == 0) {
#pragma unroll
        for (int g = 0; g < 8; ++g)
            idx_lds[(wave << 4) + (halfw << 3) + g] = bidx[g];
    }
    __syncthreads();

    // ---- Gather chosen codes (L2-resident embed) and stream outputs (NT).
    {
        const int p  = tid & 127;
        const int dp = tid >> 7;
        const int myidx = idx_lds[p];
        float* out_qwg = out;                        // quantize_with_grad (== quantize fwd)
        float* out_q   = out + (size_t)OUT_PLANE;    // quantize
#pragma unroll 4
        for (int it = 0; it < 128; ++it) {
            int d = it * 2 + dp;
            float v = embed[d * K_DIM + myidx];
            size_t o = in_base + (size_t)d * HW + p;
            __builtin_nontemporal_store(v, &out_qwg[o]);
            __builtin_nontemporal_store(v, &out_q[o]);
        }
        if (tid < 128) {
            float* out_idx = out + (size_t)(2 * (size_t)OUT_PLANE);
            out_idx[n0 + tid] = (float)idx_lds[tid];
        }
    }
}

extern "C" void kernel_launch(void* const* d_in, const int* in_sizes, int n_in,
                              void* d_out, int out_size, void* d_ws, size_t ws_size,
                              hipStream_t stream) {
    const float* input = (const float*)d_in[0];   // [32,256,64,64] f32
    const float* embed = (const float*)d_in[1];   // [256,1024] f32
    float* out = (float*)d_out;

    bf16_t* bprep = (bf16_t*)d_ws;                                       // 512 KB
    float*  enorm = (float*)((char*)d_ws + (size_t)D_DIM * K_DIM * 2);   // +4 KB

    prep_b_kernel<<<64, 256, 0, stream>>>(embed, bprep);
    enorm_kernel<<<4, 256, 0, stream>>>(embed, enorm);
    vq_main<<<N_TOTAL / N_TILE, 256, 0, stream>>>(input, embed, bprep, enorm, out);
}
